// Decoder_35828617183295
// MI455X (gfx1250) — matrix operand order repaired and checked by reading, not compiled, not run on hardware
//
#include <hip/hip_runtime.h>
#include <hip/hip_bf16.h>
#include <math.h>

// Problem constants (match reference): B=128, T=512, H=1024, C=512
#define BB 128
#define TT 512
#define HH 1024
#define CC 512

// LDS row padding (bf16 elems): row strides of 1040B / 2064B = 260 / 516
// dwords -> 4 mod 64 banks, so the 16 rows of a fragment land on distinct
// bank groups (no 16-way conflicts on ds_load_b128).
#define YPAD 520   // CC + 8
#define SPAD 1032  // HH + 8

typedef __bf16 bf16_t;
typedef __attribute__((ext_vector_type(16))) __bf16 v16bf;
typedef __attribute__((ext_vector_type(8)))  __bf16 v8bf;
typedef __attribute__((ext_vector_type(8)))  float  v8f;

#define V8F_ZERO {0.f, 0.f, 0.f, 0.f, 0.f, 0.f, 0.f, 0.f}

// ---------------------------------------------------------------------------
// Async global->LDS staging (GLOBAL_LOAD_ASYNC_TO_LDS_B128, ASYNCcnt).
// Generic LDS addresses truncate to the wave-logical LDS byte address
// (ISA 10.2 aperture rules), so the low 32 bits feed VDST directly.
// ---------------------------------------------------------------------------
__device__ __forceinline__ void async_b128_to_lds(const bf16_t* src, bf16_t* lds_dst) {
  unsigned lds_off = (unsigned)(size_t)(void*)lds_dst;
  asm volatile("global_load_async_to_lds_b128 %0, %1, off"
               :: "v"(lds_off), "v"(src) : "memory");
}
__device__ __forceinline__ void wait_async_lds() {
  asm volatile("s_wait_asynccnt 0x0" ::: "memory");
}

// Stage a 16-row x ncols bf16 panel (global, row stride ld) into LDS with
// row stride pad. 256 threads, 16B per lane per issue; chunk count is a
// multiple of 256 for all our panels (no divergence).
__device__ __forceinline__ void stage16(const bf16_t* __restrict__ g, long ld,
                                        int m_base, int ncols, int pad,
                                        bf16_t* dst) {
  const int cpr = ncols >> 3;          // 16B chunks per row
  const int chunks = cpr << 4;         // 16 rows
  for (int c = threadIdx.x; c < chunks; c += 256) {
    const int row = c / cpr;
    const int col = (c - row * cpr) << 3;
    async_b128_to_lds(g + (long)(m_base + row) * ld + col, dst + row * pad + col);
  }
}

// ---------------------------------------------------------------------------
// Fragment loaders (wave32, v_wmma_f32_16x16x32_bf16 layouts per ISA 7.12.2)
// ---------------------------------------------------------------------------

// A fragment 16x32 from an LDS panel (row stride pad). lane&15 = row;
// lanes<16 hold K {0..7,16..23}, lanes>=16 hold K {8..15,24..31}.
__device__ __forceinline__ v16bf load_a_sh(const bf16_t* s, int pad,
                                           int k_base, int lane) {
  const int row = lane & 15;
  const int off = (lane >> 4) << 3;
  const bf16_t* p = s + row * pad + k_base + off;
  v8bf lo = *(const v8bf*)(p);
  v8bf hi = *(const v8bf*)(p + 16);
  return __builtin_shufflevector(lo, hi, 0, 1, 2, 3, 4, 5, 6, 7,
                                 8, 9, 10, 11, 12, 13, 14, 15);
}

// A fragment straight from global (used by the one-shot init GEMMs).
__device__ __forceinline__ v16bf load_a(const bf16_t* __restrict__ A, long lda,
                                        int m_base, int k_base, int lane) {
  const int row = m_base + (lane & 15);
  const int off = (lane >> 4) << 3;
  const bf16_t* p = A + (long)row * lda + k_base + off;
  v8bf lo = *(const v8bf*)(p);
  v8bf hi = *(const v8bf*)(p + 16);
  return __builtin_shufflevector(lo, hi, 0, 1, 2, 3, 4, 5, 6, 7,
                                 8, 9, 10, 11, 12, 13, 14, 15);
}

// B fragment 32x16 from transposed weight BT [N x K] row-major.
// lane&15 = column; lanes<16 hold K {0..7,16..23}, lanes>=16 hold K {8..15,24..31}.
__device__ __forceinline__ v16bf load_b(const bf16_t* __restrict__ BT, int K,
                                        int n_base, int k_base, int lane) {
  const int col  = n_base + (lane & 15);
  const int koff = k_base + ((lane >> 4) << 3);
  const bf16_t* p = BT + (long)col * K + koff;
  v8bf lo = *(const v8bf*)(p);
  v8bf hi = *(const v8bf*)(p + 16);
  return __builtin_shufflevector(lo, hi, 0, 1, 2, 3, 4, 5, 6, 7,
                                 8, 9, 10, 11, 12, 13, 14, 15);
}

__device__ __forceinline__ v8f wmma_bf16(v16bf a, v16bf b, v8f c) {
  return __builtin_amdgcn_wmma_f32_16x16x32_bf16(false, a, false, b,
                                                 (short)0, c, false, false);
}

__device__ __forceinline__ float sigmoidf_(float x) {
  return 1.0f / (1.0f + __expf(-x));
}

// ---------------------------------------------------------------------------
// Output-projection body (shared by the fused step kernel and the tail):
//   D[b, t, :] = s @ W_o + b_o.  b in [0,16): 8 mtiles x 2 blocks of 8 npairs.
// ---------------------------------------------------------------------------
__device__ __forceinline__ void out_body(const bf16_t* __restrict__ S,
                                         const bf16_t* __restrict__ WoT,
                                         const float* __restrict__ bo,
                                         float* __restrict__ D, long ldd,
                                         int b, int wave, int lane,
                                         bf16_t* sS) {
  const int m_base = (b >> 1) << 4;
  stage16(S, HH, m_base, HH, SPAD, sS);
  wait_async_lds();
  __syncthreads();

  const int np = ((b & 1) << 3) + wave;
  const int n0 = np << 5, n1 = n0 + 16;
  v8f a0 = V8F_ZERO, a1 = V8F_ZERO;
#pragma unroll 2
  for (int k = 0; k < HH; k += 32) {
    v16bf a = load_a_sh(sS, SPAD, k, lane);
    a0 = wmma_bf16(a, load_b(WoT, HH, n0, k, lane), a0);
    a1 = wmma_bf16(a, load_b(WoT, HH, n1, k, lane), a1);
  }
  const int cl = lane & 15;
  const int rowoff = (lane >> 4) << 3;
#pragma unroll
  for (int i = 0; i < 8; ++i) {
    const long row = m_base + rowoff + i;
    D[row * ldd + n0 + cl] = a0[i] + bo[n0 + cl];
    D[row * ldd + n1 + cl] = a1[i] + bo[n1 + cl];
  }
}

// ---------------------------------------------------------------------------
// Fused per-step kernel #1 (48 blocks):
//   blocks [0,32):  z  = sigmoid(y@Wz + s@Uz + cz) -> Z (f32)
//                   rs = sigmoid(y@Wr + s@Ur + cr) * s -> RS (bf16)
//                   (one LDS-staged A fragment feeds 4 WMMAs per K-step)
//   blocks [32,48): out[:,t-1,:] = s@Wo + b_o  (independent of z/r; fusing it
//                   here halves the sequential launch count of the scan)
// ---------------------------------------------------------------------------
__global__ __launch_bounds__(256) void k_zr_out(
    const bf16_t* __restrict__ Y, long ldy,
    const bf16_t* __restrict__ S,
    const bf16_t* __restrict__ WzT, const bf16_t* __restrict__ UzT,
    const bf16_t* __restrict__ WrT, const bf16_t* __restrict__ UrT,
    const float* __restrict__ cz, const float* __restrict__ cr,
    float* __restrict__ Z, bf16_t* __restrict__ RS,
    const bf16_t* __restrict__ WoT, const float* __restrict__ bo,
    float* __restrict__ OutD, long ldo) {
  __shared__ bf16_t sY[16 * YPAD];
  __shared__ bf16_t sS[16 * SPAD];
  const int wave = (int)(threadIdx.x >> 5);
  const int lane = (int)(threadIdx.x & 31);

  if (blockIdx.x >= 32) {  // ---- output projection of the previous step ----
    if (OutD == nullptr) return;
    out_body(S, WoT, bo, OutD, ldo, (int)blockIdx.x - 32, wave, lane, sS);
    return;
  }

  // ---- z/r gates ----
  const int m_base = ((int)blockIdx.x >> 2) << 4;  // 8 mtiles, 4 blocks each
  stage16(Y, ldy, m_base, CC, YPAD, sY);
  stage16(S, HH, m_base, HH, SPAD, sS);
  wait_async_lds();
  __syncthreads();

  const int np = (((int)blockIdx.x & 3) << 3) + wave;  // 32 npairs
  const int n0 = np << 5, n1 = n0 + 16;

  v8f az0 = V8F_ZERO, az1 = V8F_ZERO, ar0 = V8F_ZERO, ar1 = V8F_ZERO;

#pragma unroll 2
  for (int k = 0; k < CC; k += 32) {
    v16bf a = load_a_sh(sY, YPAD, k, lane);
    az0 = wmma_bf16(a, load_b(WzT, CC, n0, k, lane), az0);
    az1 = wmma_bf16(a, load_b(WzT, CC, n1, k, lane), az1);
    ar0 = wmma_bf16(a, load_b(WrT, CC, n0, k, lane), ar0);
    ar1 = wmma_bf16(a, load_b(WrT, CC, n1, k, lane), ar1);
  }
#pragma unroll 2
  for (int k = 0; k < HH; k += 32) {
    v16bf a = load_a_sh(sS, SPAD, k, lane);
    az0 = wmma_bf16(a, load_b(UzT, HH, n0, k, lane), az0);
    az1 = wmma_bf16(a, load_b(UzT, HH, n1, k, lane), az1);
    ar0 = wmma_bf16(a, load_b(UrT, HH, n0, k, lane), ar0);
    ar1 = wmma_bf16(a, load_b(UrT, HH, n1, k, lane), ar1);
  }

  // C/D layout: VGPR i -> M = i (+8 for lanes 16..31), N = lane&15.
  const int cl = lane & 15;
  const int rowoff = (lane >> 4) << 3;
#pragma unroll
  for (int i = 0; i < 8; ++i) {
    const int lrow = rowoff + i;
    const long row = m_base + lrow;
    const long i0 = row * HH + n0 + cl;
    const long i1 = row * HH + n1 + cl;
    Z[i0] = sigmoidf_(az0[i] + cz[i0]);
    Z[i1] = sigmoidf_(az1[i] + cz[i1]);
    RS[i0] = (bf16_t)(sigmoidf_(ar0[i] + cr[i0]) * (float)sS[lrow * SPAD + n0 + cl]);
    RS[i1] = (bf16_t)(sigmoidf_(ar1[i] + cr[i1]) * (float)sS[lrow * SPAD + n1 + cl]);
  }
}

// ---------------------------------------------------------------------------
// Per-step kernel #2 (32 blocks):
//   sh    = tanh(y@W + rs@U + cc)
//   s_new = (1-z)*s + z*sh  -> states[t] (f32 output) and S (bf16 staging)
// ---------------------------------------------------------------------------
__global__ __launch_bounds__(256) void k_snew(
    const bf16_t* __restrict__ Y, long ldy,
    const bf16_t* __restrict__ RS,
    const bf16_t* __restrict__ WT, const bf16_t* __restrict__ UT,
    const float* __restrict__ cc, const float* __restrict__ Z,
    bf16_t* __restrict__ S, float* __restrict__ Dstates) {
  __shared__ bf16_t sY[16 * YPAD];
  __shared__ bf16_t sR[16 * SPAD];
  const int wave = (int)(threadIdx.x >> 5);
  const int lane = (int)(threadIdx.x & 31);
  const int m_base = ((int)blockIdx.x >> 2) << 4;

  stage16(Y, ldy, m_base, CC, YPAD, sY);
  stage16(RS, HH, m_base, HH, SPAD, sR);
  wait_async_lds();
  __syncthreads();

  const int np = (((int)blockIdx.x & 3) << 3) + wave;
  const int n0 = np << 5, n1 = n0 + 16;

  v8f a0 = V8F_ZERO, a1 = V8F_ZERO;
#pragma unroll 2
  for (int k = 0; k < CC; k += 32) {
    v16bf a = load_a_sh(sY, YPAD, k, lane);
    a0 = wmma_bf16(a, load_b(WT, CC, n0, k, lane), a0);
    a1 = wmma_bf16(a, load_b(WT, CC, n1, k, lane), a1);
  }
#pragma unroll 2
  for (int k = 0; k < HH; k += 32) {
    v16bf a = load_a_sh(sR, SPAD, k, lane);
    a0 = wmma_bf16(a, load_b(UT, HH, n0, k, lane), a0);
    a1 = wmma_bf16(a, load_b(UT, HH, n1, k, lane), a1);
  }

  const int cl = lane & 15;
  const int rowoff = (lane >> 4) << 3;
#pragma unroll
  for (int i = 0; i < 8; ++i) {
    const long row = m_base + rowoff + i;
    const long i0 = row * HH + n0 + cl;
    const long i1 = row * HH + n1 + cl;
    {
      const float sh = tanhf(a0[i] + cc[i0]);
      const float z = Z[i0], sp = (float)S[i0];
      const float sn = (1.0f - z) * sp + z * sh;
      Dstates[i0] = sn; S[i0] = (bf16_t)sn;  // same-thread read->write on S
    }
    {
      const float sh = tanhf(a1[i] + cc[i1]);
      const float z = Z[i1], sp = (float)S[i1];
      const float sn = (1.0f - z) * sp + z * sh;
      Dstates[i1] = sn; S[i1] = (bf16_t)sn;
    }
  }
}

// Tail: output projection for the final step.
__global__ __launch_bounds__(256) void k_out_tail(
    const bf16_t* __restrict__ S, const bf16_t* __restrict__ WoT,
    const float* __restrict__ bo, float* __restrict__ D, long ldd) {
  __shared__ bf16_t sS[16 * SPAD];
  out_body(S, WoT, bo, D, ldd, (int)blockIdx.x,
           (int)(threadIdx.x >> 5), (int)(threadIdx.x & 31), sS);
}

// ---------------------------------------------------------------------------
// Init GEMM: D = epi(lh @ B),  M=B=128, N=K=H=1024 (runs 4x at startup).
//   epi 0: store f32 (cz/cr/cc);  epi 2: tanh -> bf16 S (s0)
// ---------------------------------------------------------------------------
__global__ __launch_bounds__(256) void k_init(
    const bf16_t* __restrict__ A, const bf16_t* __restrict__ BT,
    float* __restrict__ Df32, bf16_t* __restrict__ Dbf, int epi) {
  const int wave = (int)(threadIdx.x >> 5);
  const int lane = (int)(threadIdx.x & 31);
  const int wid  = (int)blockIdx.x * 8 + wave;
  const int m_base = (wid >> 5) << 4;
  const int n0 = (wid & 31) << 5;
  const int n1 = n0 + 16;

  v8f a0 = V8F_ZERO, a1 = V8F_ZERO;
#pragma unroll 2
  for (int k = 0; k < HH; k += 32) {
    v16bf a = load_a(A, HH, m_base, k, lane);
    a0 = wmma_bf16(a, load_b(BT, HH, n0, k, lane), a0);
    a1 = wmma_bf16(a, load_b(BT, HH, n1, k, lane), a1);
  }

  const int cl = lane & 15;
  const int rowoff = (lane >> 4) << 3;
#pragma unroll
  for (int i = 0; i < 8; ++i) {
    const long row = m_base + rowoff + i;
    const long i0 = row * HH + n0 + cl;
    const long i1 = row * HH + n1 + cl;
    if (epi == 2) {
      Dbf[i0] = (bf16_t)tanhf(a0[i]);
      Dbf[i1] = (bf16_t)tanhf(a1[i]);
    } else {
      Df32[i0] = a0[i];
      Df32[i1] = a1[i];
    }
  }
}

// ---------------------------------------------------------------------------
// Conversions
// ---------------------------------------------------------------------------
__global__ __launch_bounds__(256) void k_f32_to_bf16(
    const float* __restrict__ src, bf16_t* __restrict__ dst, long n) {
  const long i = (long)blockIdx.x * 256 + threadIdx.x;
  if (i < n) dst[i] = (bf16_t)src[i];
}

// f32 [K x N] -> bf16 [N x K] (coalesced writes)
__global__ __launch_bounds__(256) void k_transpose_convert(
    const float* __restrict__ src, bf16_t* __restrict__ dst, int K, int N) {
  const long i = (long)blockIdx.x * 256 + threadIdx.x;
  if (i >= (long)K * N) return;
  const int n = (int)(i / K);
  const int k = (int)(i - (long)n * K);
  dst[i] = (bf16_t)src[(long)k * N + n];
}

// ---------------------------------------------------------------------------
// Host-side orchestration
// ---------------------------------------------------------------------------
static inline void* carve(char*& p, size_t bytes) {
  void* r = (void*)p;
  p += (bytes + 255) & ~(size_t)255;
  return r;
}

extern "C" void kernel_launch(void* const* d_in, const int* in_sizes, int n_in,
                              void* d_out, int out_size, void* d_ws, size_t ws_size,
                              hipStream_t stream) {
  (void)in_sizes; (void)n_in; (void)out_size; (void)ws_size;

  const float* last_h = (const float*)d_in[0];
  const float* click  = (const float*)d_in[1];
  const float* W_s    = (const float*)d_in[2];
  const float* U_z    = (const float*)d_in[3];
  const float* C_z    = (const float*)d_in[4];
  const float* U_r    = (const float*)d_in[5];
  const float* C_r    = (const float*)d_in[6];
  const float* W_z    = (const float*)d_in[7];
  const float* W_r    = (const float*)d_in[8];
  const float* W_     = (const float*)d_in[9];
  const float* U_     = (const float*)d_in[10];
  const float* Cm     = (const float*)d_in[11];
  const float* W_o    = (const float*)d_in[12];
  const float* b_o    = (const float*)d_in[13];

  float* states = (float*)d_out;                          // [T,B,H]
  float* outs   = (float*)d_out + (size_t)TT * BB * HH;   // [B,T,C]

  char* p = (char*)d_ws;
  bf16_t* WsT = (bf16_t*)carve(p, (size_t)HH * HH * 2);
  bf16_t* UzT = (bf16_t*)carve(p, (size_t)HH * HH * 2);
  bf16_t* CzT = (bf16_t*)carve(p, (size_t)HH * HH * 2);
  bf16_t* UrT = (bf16_t*)carve(p, (size_t)HH * HH * 2);
  bf16_t* CrT = (bf16_t*)carve(p, (size_t)HH * HH * 2);
  bf16_t* UT  = (bf16_t*)carve(p, (size_t)HH * HH * 2);
  bf16_t* CT  = (bf16_t*)carve(p, (size_t)HH * HH * 2);
  bf16_t* WzT = (bf16_t*)carve(p, (size_t)CC * HH * 2);
  bf16_t* WrT = (bf16_t*)carve(p, (size_t)CC * HH * 2);
  bf16_t* WT  = (bf16_t*)carve(p, (size_t)CC * HH * 2);
  bf16_t* WoT = (bf16_t*)carve(p, (size_t)HH * CC * 2);
  bf16_t* Ybf = (bf16_t*)carve(p, (size_t)BB * TT * CC * 2);
  bf16_t* LH  = (bf16_t*)carve(p, (size_t)BB * HH * 2);
  bf16_t* S   = (bf16_t*)carve(p, (size_t)BB * HH * 2);
  bf16_t* RS  = (bf16_t*)carve(p, (size_t)BB * HH * 2);
  float* cz  = (float*)carve(p, (size_t)BB * HH * 4);
  float* cr  = (float*)carve(p, (size_t)BB * HH * 4);
  float* cc  = (float*)carve(p, (size_t)BB * HH * 4);
  float* zb  = (float*)carve(p, (size_t)BB * HH * 4);

  // ---- Stage 0: weight transpose+convert; activation conversion -----------
  {
    const int thr = 256;
    const int gHH = (int)(((size_t)HH * HH + thr - 1) / thr);
    const int gCH = (int)(((size_t)CC * HH + thr - 1) / thr);
    k_transpose_convert<<<gHH, thr, 0, stream>>>(W_s, WsT, HH, HH);
    k_transpose_convert<<<gHH, thr, 0, stream>>>(U_z, UzT, HH, HH);
    k_transpose_convert<<<gHH, thr, 0, stream>>>(C_z, CzT, HH, HH);
    k_transpose_convert<<<gHH, thr, 0, stream>>>(U_r, UrT, HH, HH);
    k_transpose_convert<<<gHH, thr, 0, stream>>>(C_r, CrT, HH, HH);
    k_transpose_convert<<<gHH, thr, 0, stream>>>(U_,  UT,  HH, HH);
    k_transpose_convert<<<gHH, thr, 0, stream>>>(Cm,  CT,  HH, HH);
    k_transpose_convert<<<gCH, thr, 0, stream>>>(W_z, WzT, CC, HH);
    k_transpose_convert<<<gCH, thr, 0, stream>>>(W_r, WrT, CC, HH);
    k_transpose_convert<<<gCH, thr, 0, stream>>>(W_,  WT,  CC, HH);
    k_transpose_convert<<<gCH, thr, 0, stream>>>(W_o, WoT, HH, CC);

    const long ny = (long)BB * TT * CC;
    k_f32_to_bf16<<<(int)((ny + thr - 1) / thr), thr, 0, stream>>>(click, Ybf, ny);
    const long nh = (long)BB * HH;
    k_f32_to_bf16<<<(int)((nh + thr - 1) / thr), thr, 0, stream>>>(last_h, LH, nh);
  }

  // ---- Stage 1: init.  s0 = tanh(lh@W_s) -> S(bf16); c* = lh@C_* (f32) ----
  k_init<<<32, 256, 0, stream>>>(LH, WsT, nullptr, S, 2);
  k_init<<<32, 256, 0, stream>>>(LH, CzT, cz, nullptr, 0);
  k_init<<<32, 256, 0, stream>>>(LH, CrT, cr, nullptr, 0);
  k_init<<<32, 256, 0, stream>>>(LH, CT,  cc, nullptr, 0);

  // ---- Stage 2: recurrent scan, 2 launches per step -----------------------
  const long ldy = (long)TT * CC;
  for (int t = 0; t < TT; ++t) {
    const bf16_t* y_t = Ybf + (size_t)t * CC;
    float* out_prev = (t == 0) ? nullptr : outs + (size_t)(t - 1) * CC;
    k_zr_out<<<48, 256, 0, stream>>>(y_t, ldy, S, WzT, UzT, WrT, UrT,
                                     cz, cr, zb, RS, WoT, b_o, out_prev, ldy);
    k_snew<<<32, 256, 0, stream>>>(y_t, ldy, RS, WT, UT, cc, zb, S,
                                   states + (size_t)t * BB * HH);
  }
  // Tail: output projection of the final step.
  k_out_tail<<<16, 256, 0, stream>>>(S, WoT, b_o, outs + (size_t)(TT - 1) * CC, ldy);
}
